// PixelShuffleLIFBlock_11991548690661
// MI455X (gfx1250) — compile-verified, matrix operand order checked by reading, not run on hardware
//
#include <hip/hip_runtime.h>

typedef __attribute__((ext_vector_type(16))) _Float16 v16h;
typedef __attribute__((ext_vector_type(8)))  _Float16 v8h;
typedef __attribute__((ext_vector_type(8)))  float    v8f;
typedef __attribute__((ext_vector_type(4)))  float    v4f;

#define T_    4
#define B_    8
#define C_    64
#define H_    128
#define W_    128
#define HS    64
#define WS    64
#define KK    256      // C * r^2
#define DECAY 0.25f

// Fused LIF (16 virtual timesteps) + 1x1 conv at LOW resolution via WMMA.
// One block = one (batch b, 8x8 low-res spatial tile). Spikes never leave LDS.
__global__ __launch_bounds__(256)
void lif_conv_kernel(const float* __restrict__ x,
                     const float* __restrict__ w,
                     float* __restrict__ Y)
{
    __shared__ _Float16 Wlds[C_ * KK];   // [c_out][k]   32 KB
    __shared__ _Float16 St[64 * KK];     // [s][k]       32 KB (k contiguous -> 16B frag loads)

    const int tid  = threadIdx.x;
    const int lane = tid & 31;
    const int wv   = tid >> 5;

    const int b    = blockIdx.x >> 6;
    const int tile = blockIdx.x & 63;
    const int hs0  = (tile >> 3) * 8;
    const int ws0  = (tile & 7) * 8;

    // Stage f16 weights in LDS once (A-matrix of every GEMM).
    for (int i = tid; i < C_ * KK; i += 256) Wlds[i] = (_Float16)w[i];

    // LIF state: this thread owns 16 channels at one spatial cell.
    const int s     = tid & 63;           // 0..63  (th*8+tw)
    const int cbase = (tid >> 6) << 4;    // 0,16,32,48
    const int th = s >> 3, tw = s & 7;
    const int hrow = (hs0 + th) * 2;      // high-res row of phase i=0
    const int wcol = (ws0 + tw) * 2;      // high-res col of phase j=0

    float mem[16], spk[16];
    #pragma unroll
    for (int i = 0; i < 16; ++i) { mem[i] = 0.f; spk[i] = 0.f; }

    __syncthreads();

    for (int t = 0; t < T_; ++t) {
        const float* xt = x + (size_t)(t * B_ + b) * C_ * H_ * W_;

        // ---- LIF over the 4 pixel-unshuffle phases (sequential scan) ----
        // One float2 load covers both j phases of a row: dense 256B/wave stream.
        #pragma unroll
        for (int i2 = 0; i2 < 2; ++i2) {
            #pragma unroll
            for (int cc = 0; cc < 16; ++cc) {
                const int c = cbase + cc;
                const float2 xv = *(const float2*)
                    &xt[((size_t)c * H_ + hrow + i2) * W_ + wcol];
                // phase p = i2*2 + 0
                float m0 = (mem[cc] - spk[cc]) * DECAY + xv.x;
                float q0 = rintf(fminf(fmaxf(m0, 0.f), 4.f)) * 0.25f;
                // phase p = i2*2 + 1
                float m1 = (m0 - q0) * DECAY + xv.y;
                float q1 = rintf(fminf(fmaxf(m1, 0.f), 4.f)) * 0.25f;
                mem[cc] = m1;
                spk[cc] = q1;
                St[s * KK + c * 4 + i2 * 2 + 0] = (_Float16)q0;
                St[s * KK + c * 4 + i2 * 2 + 1] = (_Float16)q1;
            }
        }
        __syncthreads();

        // Prefetch next timestep's x tile while the GEMM runs (global_prefetch_b8).
        if (t + 1 < T_) {
            const float* nx = x + (size_t)((t + 1) * B_ + b) * C_ * H_ * W_;
            #pragma unroll
            for (int cc = 0; cc < 16; cc += 4) {
                __builtin_prefetch(
                    &nx[((size_t)(cbase + cc) * H_ + hrow) * W_ + wcol], 0, 1);
            }
        }

        // ---- GEMM: Y[c_out, s] = sum_k W[c_out,k] * St[s,k] ----
        // 16 output tiles of 16x16; each wave owns 2 N-adjacent tiles that share
        // the same A fragment -> load A once per kc, feed two WMMAs.
        const int lh  = lane & 15, sel = lane >> 4;
        const int mt  = wv >> 1;          // both of this wave's tiles share mt
        const int nt0 = (wv & 1) * 2;     // tiles nt0 and nt0+1

        v8f acc0 = {}, acc1 = {};
        #pragma unroll
        for (int kc = 0; kc < 8; ++kc) {
            // 16-bit A layout: lanes 0-15 hold K {0..7,16..23}, lanes 16-31 hold {8..15,24..31}
            const int k0 = kc * 32 + sel * 8;
            const int ra = (mt * 16 + lh) * KK + k0;
            v8h alo = *(const v8h*)&Wlds[ra];
            v8h ahi = *(const v8h*)&Wlds[ra + 16];
            v16h a = __builtin_shufflevector(alo, ahi,
                     0,1,2,3,4,5,6,7,8,9,10,11,12,13,14,15);

            const int rb0 = ((nt0 + 0) * 16 + lh) * KK + k0;
            v8h b0lo = *(const v8h*)&St[rb0];
            v8h b0hi = *(const v8h*)&St[rb0 + 16];
            v16h bf0 = __builtin_shufflevector(b0lo, b0hi,
                       0,1,2,3,4,5,6,7,8,9,10,11,12,13,14,15);

            const int rb1 = ((nt0 + 1) * 16 + lh) * KK + k0;
            v8h b1lo = *(const v8h*)&St[rb1];
            v8h b1hi = *(const v8h*)&St[rb1 + 16];
            v16h bf1 = __builtin_shufflevector(b1lo, b1hi,
                       0,1,2,3,4,5,6,7,8,9,10,11,12,13,14,15);

            acc0 = __builtin_amdgcn_wmma_f32_16x16x32_f16(
                       false, a, false, bf0, (short)0, acc0, false, false);
            acc1 = __builtin_amdgcn_wmma_f32_16x16x32_f16(
                       false, a, false, bf1, (short)0, acc1, false, false);
        }

        // C/D layout: VGPR r -> M = r + 8*sel, N = lane&15
        float* Yt = Y + (size_t)(t * B_ + b) * C_ * HS * WS;
        #pragma unroll
        for (int r = 0; r < 8; ++r) {
            const int co = mt * 16 + r + 8 * sel;
            const int sc0 = (nt0 + 0) * 16 + lh;
            const int sc1 = (nt0 + 1) * 16 + lh;
            Yt[((size_t)co * HS + hs0 + (sc0 >> 3)) * WS + ws0 + (sc0 & 7)] = acc0[r];
            Yt[((size_t)co * HS + hs0 + (sc1 >> 3)) * WS + ws0 + (sc1 & 7)] = acc1[r];
        }
        __syncthreads();   // St reads done before next t's LIF writes
    }
}

// Bilinear 2x upsample (half-pixel centers, edge clamp). Each thread produces
// a 4-float quad (4 consecutive output pixels) from 4 source columns x 2 source
// rows. Output is write-once -> non-temporal b128 stores keep L2 free for Y.
__global__ __launch_bounds__(256)
void upsample2x_kernel(const float* __restrict__ Y, float* __restrict__ out)
{
    const int plane = blockIdx.y;                       // (t*B+b)*C + c : 0..2047
    const int qi    = blockIdx.x * 256 + threadIdx.x;   // 0..4095 quads
    const int h  = qi >> 5;        // output row 0..127
    const int q  = qi & 31;        // quad in row
    const int wb = q * 2;          // central source col pair: wb, wb+1

    // vertical taps
    const int e = h >> 1;
    int r0, r1; float wh0, wh1;
    if (h & 1) { r0 = e; r1 = (e + 1 < HS) ? e + 1 : HS - 1; wh0 = 0.75f; wh1 = 0.25f; }
    else       { r0 = (e > 0) ? e - 1 : 0; r1 = e;           wh0 = 0.25f; wh1 = 0.75f; }

    const int cm1 = (wb > 0) ? wb - 1 : 0;
    const int cp2 = (wb + 2 < WS) ? wb + 2 : WS - 1;

    const float* Yp = Y + (size_t)plane * HS * WS;
    const float* ra = Yp + r0 * WS;
    const float* rb = Yp + r1 * WS;

    const float s0 = wh0 * ra[cm1]    + wh1 * rb[cm1];
    const float s1 = wh0 * ra[wb]     + wh1 * rb[wb];
    const float s2 = wh0 * ra[wb + 1] + wh1 * rb[wb + 1];
    const float s3 = wh0 * ra[cp2]    + wh1 * rb[cp2];

    v4f o;
    o[0] = 0.25f * s0 + 0.75f * s1;   // out col 4q   (even)
    o[1] = 0.75f * s1 + 0.25f * s2;   // out col 4q+1 (odd)
    o[2] = 0.25f * s1 + 0.75f * s2;   // out col 4q+2 (even)
    o[3] = 0.75f * s2 + 0.25f * s3;   // out col 4q+3 (odd)
    __builtin_nontemporal_store(
        o, (v4f*)&out[(size_t)plane * H_ * W_ + (size_t)h * W_ + wb * 2]);
}

extern "C" void kernel_launch(void* const* d_in, const int* in_sizes, int n_in,
                              void* d_out, int out_size, void* d_ws, size_t ws_size,
                              hipStream_t stream)
{
    const float* x = (const float*)d_in[0];   // [4,8,64,128,128] f32
    const float* w = (const float*)d_in[1];   // [64,256] f32
    float* out = (float*)d_out;               // [4,8,64,128,128] f32
    float* Y   = (float*)d_ws;                // [4,8,64,64,64]   f32 (33.5 MB scratch)

    lif_conv_kernel<<<B_ * 64, 256, 0, stream>>>(x, w, Y);
    upsample2x_kernel<<<dim3((H_ * W_) / (4 * 256), T_ * B_ * C_), 256, 0, stream>>>(Y, out);
}
